// GPT_41781441856168
// MI455X (gfx1250) — compile-verified, matrix operand order checked by reading, not run on hardware
//
#include <hip/hip_runtime.h>
#include <math.h>

// ---------------- problem constants (GPT-2 124M, B=2, T=1024) ----------------
#define V_   50257
#define C_   768
#define L_   12
#define H_   12
#define T_   1024
#define B_   2
#define BT_  (B_ * T_)      // 2048 token rows
#define HD_  64
#define C3_  (3 * C_)       // 2304
#define C4_  (4 * C_)       // 3072

// ---------------- WMMA types ----------------
typedef __bf16 v16bf __attribute__((ext_vector_type(16)));
typedef float  v8f   __attribute__((ext_vector_type(8)));

union FragB16 { v16bf v; unsigned int u[8]; };

__device__ __forceinline__ unsigned short f2bf(float f) {
    // round-to-nearest-even fp32 -> bf16
    unsigned int u = __builtin_bit_cast(unsigned int, f);
    return (unsigned short)((u + 0x7FFFu + ((u >> 16) & 1u)) >> 16);
}
__device__ __forceinline__ unsigned int pack2bf(float lo, float hi) {
    return (unsigned int)f2bf(lo) | ((unsigned int)f2bf(hi) << 16);
}
__device__ __forceinline__ float gelu_tanh(float x) {
    float x3 = x * x * x;
    return 0.5f * x * (1.0f + tanhf(0.7978845608028654f * (x + 0.044715f * x3)));
}

// ---------------- generic WMMA bf16 GEMM ----------------
// out = act(A[M,K](bf16) @ B + bias) + resid
// bmode: 0 = B fp32 [K,N] ld=ldb ; 1 = B fp32 [N,K] ld=ldb ; 2 = B bf16 [N,K] ld=ldb
// output: outb (bf16) if non-null, else outf (fp32)
#define TM 64
#define TNB 128          // block N tile
#define TK 32
#define TKP 40           // padded row (80 B): aligned b128 stores, conflict-free reads

__device__ __forceinline__ void stage_tile(
    unsigned short (*sA)[TKP], unsigned short (*sB)[TKP],
    const unsigned short* __restrict__ A, const unsigned short* __restrict__ Bbf,
    const float* __restrict__ Bf32,
    int mb, int nb, int k0, int N, int K, int ldb, int bmode, int tid)
{
    // A tile 64x32 bf16: pure 16-byte copies (1 chunk/thread)
    {
        int m = tid >> 2, koff = (tid & 3) * 8;
        *(uint4*)&sA[m][koff] =
            *(const uint4*)(A + (size_t)(mb + m) * K + (k0 + koff));
    }
    // B tile 128x32
    if (bmode == 2) {
#pragma unroll
        for (int e = 0; e < 2; ++e) {
            int cidx = e * 256 + tid;          // 512 chunks of 8 bf16
            int n = cidx >> 2, koff = (cidx & 3) * 8;
            int gn = nb + n;
            uint4 v = {0u, 0u, 0u, 0u};
            if (gn < N)
                v = *(const uint4*)(Bbf + (size_t)gn * ldb + (k0 + koff));
            *(uint4*)&sB[n][koff] = v;
        }
    } else {
        // fp32 fallback: scalar convert + transpose into LDS
#pragma unroll
        for (int e = 0; e < 16; ++e) {
            int i = e * 256 + tid;             // 4096 elements
            int k = i >> 7, n = i & 127;
            int gn = nb + n;
            float v = 0.f;
            if (gn < N)
                v = (bmode == 1) ? Bf32[(size_t)gn * ldb + (k0 + k)]
                                 : Bf32[(size_t)(k0 + k) * ldb + gn];
            sB[n][k] = f2bf(v);
        }
    }
}

__global__ __launch_bounds__(256) void gemm_wmma_bf16(
    const unsigned short* __restrict__ A, const float* __restrict__ Bf32,
    const unsigned short* __restrict__ Bbf,
    const float* __restrict__ bias, const float* __restrict__ resid,
    float* __restrict__ outf, unsigned short* __restrict__ outb,
    int M, int N, int K, int ldb, int bmode, int act)
{
    __shared__ unsigned short smA[2][TM][TKP];
    __shared__ unsigned short smB[2][TNB][TKP];

    const int tid  = threadIdx.x;
    const int mb   = blockIdx.y * TM;
    const int nb   = blockIdx.x * TNB;
    const int wave = tid >> 5;
    const int lane = tid & 31;
    const int half = lane >> 4;       // K-half select per ISA 16-bit A layout
    const int l16  = lane & 15;
    const int wm   = (wave & 3) * 16;   // 4 M-subtiles of 16
    const int wn   = (wave >> 2) * 64;  // 2 N-groups of 64 (4 wmma tiles each)

    v8f c[4];
#pragma unroll
    for (int s = 0; s < 4; ++s) c[s] = (v8f){0.f,0.f,0.f,0.f,0.f,0.f,0.f,0.f};

    const int nsteps = K / TK;
    stage_tile(smA[0], smB[0], A, Bbf, Bf32, mb, nb, 0, N, K, ldb, bmode, tid);
    int cur = 0;

    for (int s = 0; s < nsteps; ++s) {
        __syncthreads();                       // tile `cur` visible to all waves
        if (s + 1 < nsteps)                    // prefetch next tile into other buffer
            stage_tile(smA[cur ^ 1], smB[cur ^ 1], A, Bbf, Bf32,
                       mb, nb, (s + 1) * TK, N, K, ldb, bmode, tid);

        // fragments per ISA 16-bit 16x32 layout:
        //   pair p<4 -> K = 2p + 8*half ; pair p>=4 -> K = 16+2(p-4) + 8*half
        FragB16 a;
        const unsigned short* ar = &smA[cur][wm + l16][0];
#pragma unroll
        for (int p = 0; p < 8; ++p) {
            int kk = ((p < 4) ? (2 * p) : (16 + 2 * (p - 4))) + 8 * half;
            a.u[p] = *(const unsigned int*)(ar + kk);
        }
#pragma unroll
        for (int t = 0; t < 4; ++t) {
            FragB16 b;
            const unsigned short* br = &smB[cur][wn + t * 16 + l16][0];
#pragma unroll
            for (int p = 0; p < 8; ++p) {
                int kk = ((p < 4) ? (2 * p) : (16 + 2 * (p - 4))) + 8 * half;
                b.u[p] = *(const unsigned int*)(br + kk);
            }
            c[t] = __builtin_amdgcn_wmma_f32_16x16x32_bf16(
                false, a.v, false, b.v, (short)0, c[t], false, false);
        }
        cur ^= 1;
    }

    // epilogue: D layout lanes 0-15 N=lane, VGPR r -> M=r ; lanes 16-31 -> M=r+8
#pragma unroll
    for (int t = 0; t < 4; ++t) {
        int col = nb + wn + t * 16 + l16;
        if (col >= N) continue;
        float bv = bias ? bias[col] : 0.f;
#pragma unroll
        for (int r = 0; r < 8; ++r) {
            int row = mb + wm + r + 8 * half;
            float v = c[t][r] + bv;
            if (act == 1) v = gelu_tanh(v);
            if (resid) v += resid[(size_t)row * N + col];
            if (outb) outb[(size_t)row * N + col] = f2bf(v);
            else      outf[(size_t)row * N + col] = v;
        }
    }
}

// ---------------- weight convert: fp32 [K,N] -> bf16 [N,K] (LDS transpose) ----
__global__ __launch_bounds__(256) void convT_f32_bf16(
    const float* __restrict__ in, unsigned short* __restrict__ out, int K, int N)
{
    __shared__ unsigned short t[32][33];
    int kb = blockIdx.y * 32, nbase = blockIdx.x * 32;
    int tx = threadIdx.x & 31, ty = threadIdx.x >> 5;
#pragma unroll
    for (int r = 0; r < 32; r += 8) {
        int k = kb + ty + r, n = nbase + tx;
        float v = (k < K && n < N) ? in[(size_t)k * N + n] : 0.f;
        t[ty + r][tx] = f2bf(v);
    }
    __syncthreads();
#pragma unroll
    for (int r = 0; r < 32; r += 8) {
        int n = nbase + ty + r, k = kb + tx;
        if (n < N && k < K) out[(size_t)n * K + k] = t[tx][ty + r];
    }
}

// ---------------- straight convert fp32 -> bf16 (already K-major) ----------------
__global__ __launch_bounds__(256) void conv_f32_bf16(
    const float* __restrict__ in, unsigned short* __restrict__ out, long n4)
{
    long i = (long)blockIdx.x * 256 + threadIdx.x;    // float4 units
    if (i < n4) {
        const float4 v = ((const float4*)in)[i];
        uint2 p; p.x = pack2bf(v.x, v.y); p.y = pack2bf(v.z, v.w);
        *(uint2*)(out + i * 4) = p;
    }
}

// ---------------- embedding: x = wte[idx] + wpe[t] (fp32 residual stream) ----
__global__ __launch_bounds__(256) void embed_k(
    const int* __restrict__ idx, const float* __restrict__ wte,
    const float* __restrict__ wpe, float* __restrict__ x)
{
    int row = blockIdx.x;
    int t   = row & (T_ - 1);
    int tok = idx[row];
    const float* we = wte + (size_t)tok * C_;
    const float* pe = wpe + (size_t)t * C_;
    float* xr = x + (size_t)row * C_;
    for (int c = threadIdx.x; c < C_; c += 256)
        xr[c] = we[c] + pe[c];
}

// ---------------- LayerNorm (one block per row), bf16 output ----------------
__global__ __launch_bounds__(256) void layernorm_k(
    const float* __restrict__ x, const float* __restrict__ w,
    const float* __restrict__ b, unsigned short* __restrict__ o)
{
    __shared__ float red[256];
    int row = blockIdx.x;
    const float* xr = x + (size_t)row * C_;

    float s = 0.f;
    for (int c = threadIdx.x; c < C_; c += 256) s += xr[c];
    red[threadIdx.x] = s; __syncthreads();
    for (int off = 128; off > 0; off >>= 1) {
        if (threadIdx.x < off) red[threadIdx.x] += red[threadIdx.x + off];
        __syncthreads();
    }
    float mean = red[0] * (1.0f / C_);
    __syncthreads();

    s = 0.f;
    for (int c = threadIdx.x; c < C_; c += 256) {
        float d = xr[c] - mean; s += d * d;
    }
    red[threadIdx.x] = s; __syncthreads();
    for (int off = 128; off > 0; off >>= 1) {
        if (threadIdx.x < off) red[threadIdx.x] += red[threadIdx.x + off];
        __syncthreads();
    }
    float rstd = rsqrtf(red[0] * (1.0f / C_) + 1e-5f);

    unsigned short* orow = o + (size_t)row * C_;
    for (int c = threadIdx.x; c < C_; c += 256)
        orow[c] = f2bf((xr[c] - mean) * rstd * w[c] + b[c]);
}

// ---------------- causal attention: one block per (b, h, q), bf16 y ----------
__global__ __launch_bounds__(128) void attention_k(
    const float* __restrict__ qkv, unsigned short* __restrict__ y)
{
    __shared__ float qv[HD_];
    __shared__ float sc[T_];
    __shared__ float red[128];

    int bid = blockIdx.x;
    int q   = bid & (T_ - 1);
    int rem = bid >> 10;
    int h   = rem % H_;
    int b   = rem / H_;
    int tid = threadIdx.x;

    const size_t rq = (size_t)(b * T_ + q) * C3_;
    if (tid < HD_) qv[tid] = qkv[rq + h * HD_ + tid];
    __syncthreads();

    for (int k = tid; k <= q; k += 128) {
        const float* kr = qkv + (size_t)(b * T_ + k) * C3_ + C_ + h * HD_;
        float d = 0.f;
#pragma unroll 8
        for (int i = 0; i < HD_; ++i) d += qv[i] * kr[i];
        sc[k] = d * 0.125f;   // HD^-0.5
    }
    __syncthreads();

    float m = -3.0e38f;
    for (int k = tid; k <= q; k += 128) m = fmaxf(m, sc[k]);
    red[tid] = m; __syncthreads();
    for (int off = 64; off > 0; off >>= 1) {
        if (tid < off) red[tid] = fmaxf(red[tid], red[tid + off]);
        __syncthreads();
    }
    m = red[0]; __syncthreads();

    float s = 0.f;
    for (int k = tid; k <= q; k += 128) {
        float e = __expf(sc[k] - m);
        sc[k] = e; s += e;
    }
    red[tid] = s; __syncthreads();
    for (int off = 64; off > 0; off >>= 1) {
        if (tid < off) red[tid] += red[tid + off];
        __syncthreads();
    }
    float inv = 1.0f / red[0];
    __syncthreads();

    if (tid < HD_) {
        float acc = 0.f;
        for (int k = 0; k <= q; ++k)
            acc += sc[k] * qkv[(size_t)(b * T_ + k) * C3_ + 2 * C_ + h * HD_ + tid];
        y[(size_t)(b * T_ + q) * C_ + h * HD_ + tid] = f2bf(acc * inv);
    }
}

// ---------------- host orchestration ----------------
extern "C" void kernel_launch(void* const* d_in, const int* in_sizes, int n_in,
                              void* d_out, int out_size, void* d_ws, size_t ws_size,
                              hipStream_t stream) {
    (void)in_sizes; (void)n_in; (void)out_size;

    const int*   idx    = (const int*)  d_in[0];
    const float* wte    = (const float*)d_in[1];
    const float* wpe    = (const float*)d_in[2];
    const float* ln1_w  = (const float*)d_in[3];
    const float* ln1_b  = (const float*)d_in[4];
    const float* attn_w = (const float*)d_in[5];
    const float* attn_b = (const float*)d_in[6];
    const float* proj_w = (const float*)d_in[7];
    const float* proj_b = (const float*)d_in[8];
    const float* ln2_w  = (const float*)d_in[9];
    const float* ln2_b  = (const float*)d_in[10];
    const float* fc_w   = (const float*)d_in[11];
    const float* fc_b   = (const float*)d_in[12];
    const float* fcp_w  = (const float*)d_in[13];
    const float* fcp_b  = (const float*)d_in[14];
    const float* lnf_w  = (const float*)d_in[15];
    const float* lnf_b  = (const float*)d_in[16];
    float* outp = (float*)d_out;

    // fp32 activations: x0, x1 (residual stream), qkv (softmax input)
    float* ws = (float*)d_ws;
    float* x0  = ws;
    float* x1  = x0 + (size_t)BT_ * C_;
    float* qkv = x1 + (size_t)BT_ * C_;
    const size_t f32Floats = (size_t)BT_ * (2 * C_ + C3_);

    // bf16 activations: h (LN out), y (attn out), fcb (GELU out)
    unsigned short* bact = (unsigned short*)(ws + f32Floats);
    unsigned short* h   = bact;
    unsigned short* y   = h + (size_t)BT_ * C_;
    unsigned short* fcb = y + (size_t)BT_ * C_;
    const size_t bactElems = (size_t)BT_ * (2 * C_ + C4_);
    const size_t actBytes = f32Floats * 4 + bactElems * 2;

    // bf16 weight cache ([N,K] layout)
    const size_t wAttn = (size_t)C_ * C3_;
    const size_t wProj = (size_t)C_ * C_;
    const size_t wFc   = (size_t)C_ * C4_;
    const size_t wFcp  = (size_t)C4_ * C_;
    const size_t wWte  = (size_t)V_ * C_;
    const size_t wbfElems = (size_t)L_ * (wAttn + wProj + wFc + wFcp) + wWte;
    const int use_bf16 = (ws_size >= actBytes + wbfElems * 2) ? 1 : 0;

    unsigned short* wAttnB = bact + bactElems;
    unsigned short* wProjB = wAttnB + (size_t)L_ * wAttn;
    unsigned short* wFcB   = wProjB + (size_t)L_ * wProj;
    unsigned short* wFcpB  = wFcB   + (size_t)L_ * wFc;
    unsigned short* wWteB  = wFcpB  + (size_t)L_ * wFcp;

    const dim3 blk256(256), blk128(128);
    const dim3 gN3(C3_ / TNB, BT_ / TM);
    const dim3 gN1(C_  / TNB, BT_ / TM);
    const dim3 gN4(C4_ / TNB, BT_ / TM);
    const dim3 gV((V_ + TNB - 1) / TNB, BT_ / TM);

    if (use_bf16) {
        for (int l = 0; l < L_; ++l) {
            convT_f32_bf16<<<dim3(C3_/32, C_/32), blk256, 0, stream>>>(
                attn_w + (size_t)l * wAttn, wAttnB + (size_t)l * wAttn, C_, C3_);
            convT_f32_bf16<<<dim3(C_/32, C_/32), blk256, 0, stream>>>(
                proj_w + (size_t)l * wProj, wProjB + (size_t)l * wProj, C_, C_);
            convT_f32_bf16<<<dim3(C4_/32, C_/32), blk256, 0, stream>>>(
                fc_w + (size_t)l * wFc, wFcB + (size_t)l * wFc, C_, C4_);
            convT_f32_bf16<<<dim3(C_/32, C4_/32), blk256, 0, stream>>>(
                fcp_w + (size_t)l * wFcp, wFcpB + (size_t)l * wFcp, C4_, C_);
        }
        long n4 = (long)(wWte / 4);
        conv_f32_bf16<<<(unsigned)((n4 + 255) / 256), blk256, 0, stream>>>(
            wte, wWteB, n4);
    }

    embed_k<<<BT_, blk256, 0, stream>>>(idx, wte, wpe, x0);

    for (int l = 0; l < L_; ++l) {
        layernorm_k<<<BT_, blk256, 0, stream>>>(x0, ln1_w + (size_t)l * C_,
                                                ln1_b + (size_t)l * C_, h);
        if (use_bf16)
            gemm_wmma_bf16<<<gN3, blk256, 0, stream>>>(
                h, nullptr, wAttnB + (size_t)l * wAttn,
                attn_b + (size_t)l * C3_, nullptr, qkv, nullptr,
                BT_, C3_, C_, C_, 2, 0);
        else
            gemm_wmma_bf16<<<gN3, blk256, 0, stream>>>(
                h, attn_w + (size_t)l * wAttn, nullptr,
                attn_b + (size_t)l * C3_, nullptr, qkv, nullptr,
                BT_, C3_, C_, C3_, 0, 0);

        attention_k<<<B_ * H_ * T_, blk128, 0, stream>>>(qkv, y);

        if (use_bf16)
            gemm_wmma_bf16<<<gN1, blk256, 0, stream>>>(
                y, nullptr, wProjB + (size_t)l * wProj,
                proj_b + (size_t)l * C_, x0, x1, nullptr,
                BT_, C_, C_, C_, 2, 0);
        else
            gemm_wmma_bf16<<<gN1, blk256, 0, stream>>>(
                y, proj_w + (size_t)l * wProj, nullptr,
                proj_b + (size_t)l * C_, x0, x1, nullptr,
                BT_, C_, C_, C_, 0, 0);

        layernorm_k<<<BT_, blk256, 0, stream>>>(x1, ln2_w + (size_t)l * C_,
                                                ln2_b + (size_t)l * C_, h);
        if (use_bf16)
            gemm_wmma_bf16<<<gN4, blk256, 0, stream>>>(
                h, nullptr, wFcB + (size_t)l * wFc,
                fc_b + (size_t)l * C4_, nullptr, nullptr, fcb,
                BT_, C4_, C_, C_, 2, 1);
        else
            gemm_wmma_bf16<<<gN4, blk256, 0, stream>>>(
                h, fc_w + (size_t)l * wFc, nullptr,
                fc_b + (size_t)l * C4_, nullptr, nullptr, fcb,
                BT_, C4_, C_, C4_, 0, 1);

        if (use_bf16)
            gemm_wmma_bf16<<<gN1, blk256, 0, stream>>>(
                fcb, nullptr, wFcpB + (size_t)l * wFcp,
                fcp_b + (size_t)l * C_, x1, x0, nullptr,
                BT_, C_, C4_, C4_, 2, 0);
        else
            gemm_wmma_bf16<<<gN1, blk256, 0, stream>>>(
                fcb, fcp_w + (size_t)l * wFcp, nullptr,
                fcp_b + (size_t)l * C_, x1, x0, nullptr,
                BT_, C_, C4_, C_, 0, 0);
    }

    layernorm_k<<<BT_, blk256, 0, stream>>>(x0, lnf_w, lnf_b, h);
    if (use_bf16)
        gemm_wmma_bf16<<<gV, blk256, 0, stream>>>(
            h, nullptr, wWteB, nullptr, nullptr, outp, nullptr,
            BT_, V_, C_, C_, 2, 0);
    else
        gemm_wmma_bf16<<<gV, blk256, 0, stream>>>(
            h, wte, nullptr, nullptr, nullptr, outp, nullptr,
            BT_, V_, C_, C_, 1, 0);
}
